// VoxelProposalLayer_24172075942481
// MI455X (gfx1250) — compile-verified, matrix operand order checked by reading, not run on hardware
//
#include <hip/hip_runtime.h>
#include <hip/hip_bf16.h>

// ---------------- problem constants (from reference) ----------------
#define V_TOT   262144      // 128*128*16 voxels / queries
#define S_TOT   16128       // 64*192 + 32*96 + 16*48 value pixels
#define CDIM    128
#define NHEAD   8
#define NLVL    3
#define NPT     4
#define DH      16

typedef __attribute__((ext_vector_type(16))) __bf16 v16bf;
typedef __attribute__((ext_vector_type(8)))  __bf16 v8bf;
typedef __attribute__((ext_vector_type(8)))  float  v8f;

// ---------------- WMMA fragment loaders (wave32) ---------------------------
// 16-bit A 16x32 layout (CDNA5 ISA 7.12.2): for lane (half, m),
// fragment elems i=0..7  -> K = k0 + half*8 + i         (contiguous run)
//               i=8..15 -> K = k0 + 16 + half*8 + (i-8) (contiguous run)
// => vectorizable as b128 loads + v_cvt_pk_bf16_f32.

template <int BASE>
__device__ __forceinline__ void frag_put8_f32(v16bf& v, const float* p) {
  const float4 lo = *(const float4*)p;
  const float4 hi = *(const float4*)(p + 4);
  v[BASE + 0] = (__bf16)lo.x; v[BASE + 1] = (__bf16)lo.y;
  v[BASE + 2] = (__bf16)lo.z; v[BASE + 3] = (__bf16)lo.w;
  v[BASE + 4] = (__bf16)hi.x; v[BASE + 5] = (__bf16)hi.y;
  v[BASE + 6] = (__bf16)hi.z; v[BASE + 7] = (__bf16)hi.w;
}

template <int BASE>
__device__ __forceinline__ void frag_put8_sum(v16bf& v, const float* p,
                                              const float* q) {
  const float4 alo = *(const float4*)p;
  const float4 ahi = *(const float4*)(p + 4);
  const float4 blo = *(const float4*)q;
  const float4 bhi = *(const float4*)(q + 4);
  v[BASE + 0] = (__bf16)(alo.x + blo.x); v[BASE + 1] = (__bf16)(alo.y + blo.y);
  v[BASE + 2] = (__bf16)(alo.z + blo.z); v[BASE + 3] = (__bf16)(alo.w + blo.w);
  v[BASE + 4] = (__bf16)(ahi.x + bhi.x); v[BASE + 5] = (__bf16)(ahi.y + bhi.y);
  v[BASE + 6] = (__bf16)(ahi.z + bhi.z); v[BASE + 7] = (__bf16)(ahi.w + bhi.w);
}

template <int BASE>
__device__ __forceinline__ void frag_put8_bf16(v16bf& v, const __bf16* p) {
  const v8bf t = *(const v8bf*)p;  // 16B vector load
#pragma unroll
  for (int i = 0; i < 8; ++i) v[BASE + i] = t[i];
}

// Load + convert the 4 K-step A fragments (full K = 128) ONCE into registers.
template <bool ADDQ>
__device__ __forceinline__ void load_a_frags(v16bf (&af)[4],
                                             const float* __restrict__ A0,
                                             const float* __restrict__ A1,
                                             int row0) {
  const int lane = threadIdx.x & 31;
  const int half = lane >> 4;
  const int m    = lane & 15;
  const float* ar = A0 + (size_t)(row0 + m) * CDIM;
  const float* qr = ADDQ ? A1 + (size_t)(row0 + m) * CDIM : nullptr;
#pragma unroll
  for (int kk = 0; kk < 4; ++kk) {
    const int r0 = kk * 32 + half * 8;
    const int r1 = r0 + 16;
    if (ADDQ) {
      frag_put8_sum<0>(af[kk], ar + r0, qr + r0);
      frag_put8_sum<8>(af[kk], ar + r1, qr + r1);
    } else {
      frag_put8_f32<0>(af[kk], ar + r0);
      frag_put8_f32<8>(af[kk], ar + r1);
    }
  }
}

// One 16x16 tile against resident A fragments; B from pre-transposed bf16
// weights WT [N][128] (contiguous K). 4 chained v_wmma_f32_16x16x32_bf16.
__device__ __forceinline__ v8f wmma_cols(const v16bf (&af)[4],
                                         const __bf16* __restrict__ WT,
                                         int col0) {
  const int lane = threadIdx.x & 31;
  const int half = lane >> 4;
  const int n    = lane & 15;
  const __bf16* br = WT + (size_t)(col0 + n) * CDIM;
  v8f c = {};
#pragma unroll
  for (int kk = 0; kk < 4; ++kk) {
    const int r0 = kk * 32 + half * 8;
    const int r1 = r0 + 16;
    v16bf b;
    frag_put8_bf16<0>(b, br + r0);
    frag_put8_bf16<8>(b, br + r1);
    c = __builtin_amdgcn_wmma_f32_16x16x32_bf16(
        /*neg_a=*/false, af[kk], /*neg_b=*/false, b,
        /*c_mod=*/(short)0, c, /*reuse_a=*/false, /*reuse_b=*/false);
  }
  return c;
}

// C/D tile layout: lane (half, n), VGPR r -> element (m = r + 8*half, n).
// Compile-time row stride N => per-lane base pointer + immediate offsets
// (r*N*4 fits the 24-bit signed IOFFSET), no per-store address chain.
template <int N>
__device__ __forceinline__ void store_tile_bias(const v8f& c,
                                                float* __restrict__ out,
                                                const float* __restrict__ bias,
                                                int row0, int col0) {
  const int lane = threadIdx.x & 31;
  const int half = lane >> 4;
  const int n    = lane & 15;
  const float bv = bias[col0 + n];
  float* p = out + (size_t)(row0 + 8 * half) * N + col0 + n;
#pragma unroll
  for (int r = 0; r < 8; ++r) p[r * N] = c[r] + bv;
}

// ---------------- kernel 0: pts_mask (init + scatter) ----------------------
__global__ void vpl_mask_init(int* __restrict__ mask) {
  const int i = blockIdx.x * blockDim.x + threadIdx.x;
  if (i < V_TOT) mask[i] = 0;
}

__global__ void vpl_mask_scatter(const int* __restrict__ vol_pts,
                                 int* __restrict__ mask) {
  const int i = blockIdx.x * blockDim.x + threadIdx.x;
  if (i >= V_TOT) return;
  const int x = vol_pts[3 * i + 0];
  const int y = vol_pts[3 * i + 1];
  const int z = vol_pts[3 * i + 2];
  if (x >= 0 && x < 128 && y >= 0 && y < 128 && z >= 0 && z < 16)
    atomicOr(&mask[x * 2048 + y * 16 + z], 1);
}

// ---------------- kernel 1: concat multi-scale feats -> value[S,128] -------
__global__ void vpl_build_value(const float* __restrict__ f0,
                                const float* __restrict__ f1,
                                const float* __restrict__ f2,
                                float* __restrict__ value) {
  const int idx = blockIdx.x * blockDim.x + threadIdx.x;
  if (idx >= S_TOT * CDIM) return;
  const int s = idx >> 7, c = idx & 127;
  const float* f; int p, hw;
  if (s < 12288)       { f = f0; p = s;         hw = 12288; }
  else if (s < 15360)  { f = f1; p = s - 12288; hw = 3072;  }
  else                 { f = f2; p = s - 15360; hw = 768;   }
  value[idx] = f[c * hw + p];
}

// ---------------- kernel 2: weight transpose + f32->bf16 -------------------
// W [K=128, N] row-major f32  ->  WT [N, 128] bf16 (contiguous K per column)
__global__ void vpl_weight_bf16(const float* __restrict__ W,
                                __bf16* __restrict__ WT, int N) {
  const int idx = blockIdx.x * blockDim.x + threadIdx.x;
  if (idx >= N * CDIM) return;
  const int n = idx >> 7, k = idx & 127;
  WT[idx] = (__bf16)W[k * N + n];
}

// ---------------- kernel 3: value projection GEMM (A is L2-resident) -------
__global__ __launch_bounds__(32)
void vpl_gemm_vproj(const float* __restrict__ A, const __bf16* __restrict__ WT,
                    const float* __restrict__ bias, float* __restrict__ out) {
  const int row0 = blockIdx.x * 16;
  const int col0 = blockIdx.y * 16;
  v16bf af[4];
  load_a_frags<false>(af, A, nullptr, row0);
  const v8f c = wmma_cols(af, WT, col0);
  store_tile_bias<CDIM>(c, out, bias, row0, col0);
}

// ---------------- kernel 4: fused offset + attention-logit GEMM ------------
// One wave owns a full 16-row panel: q = se+sp fragments are loaded/converted
// ONCE (HBM traffic = 1x 268 MB), then swept over all 18 column tiles
// (12 offset tiles + 6 logit tiles) against the L2-resident bf16 weights.
__global__ __launch_bounds__(32)
void vpl_gemm_off_attn(const float* __restrict__ se,
                       const float* __restrict__ sp,
                       const __bf16* __restrict__ wt_off,
                       const float* __restrict__ boff,
                       const __bf16* __restrict__ wt_attn,
                       const float* __restrict__ battn,
                       float* __restrict__ off, float* __restrict__ logit) {
  const int row0 = blockIdx.x * 16;
  v16bf af[4];
  load_a_frags<true>(af, se, sp, row0);
#pragma unroll
  for (int t = 0; t < 12; ++t) {
    const int col0 = t * 16;
    const v8f c = wmma_cols(af, wt_off, col0);
    store_tile_bias<192>(c, off, boff, row0, col0);
  }
#pragma unroll
  for (int t = 0; t < 6; ++t) {
    const int col0 = t * 16;
    const v8f c = wmma_cols(af, wt_attn, col0);
    store_tile_bias<96>(c, logit, battn, row0, col0);
  }
}

// ---------------- kernel 5: softmax + bilinear deformable sampling ---------
// block = 1 voxel, 128 threads = (head h = t>>4, channel d = t&15).
__global__ __launch_bounds__(128)
void vpl_sample(const float* __restrict__ ref_pix,
                const float* __restrict__ off,
                const float* __restrict__ logit,
                const float* __restrict__ vproj,
                float* __restrict__ acc) {
  const int v = blockIdx.x;
  const int t = threadIdx.x;
  const int h = t >> 4;
  const int d = t & 15;

  if (v + 1 < V_TOT)  // speculative prefetch of next voxel's offsets
    __builtin_prefetch(off + (size_t)(v + 1) * 192, 0, 1);

  const float rx = ref_pix[2 * v + 0];
  const float ry = ref_pix[2 * v + 1];

  // in-register softmax over the 12 (level,point) logits for this head
  float lg[NLVL * NPT];
  const float* L = logit + (size_t)v * 96 + h * 12;
  float mx = -1e30f;
#pragma unroll
  for (int i = 0; i < 12; ++i) { lg[i] = L[i]; mx = fmaxf(mx, lg[i]); }
  float sum = 0.f;
#pragma unroll
  for (int i = 0; i < 12; ++i) { lg[i] = __expf(lg[i] - mx); sum += lg[i]; }
  const float inv = 1.f / sum;

  const float* O = off + (size_t)v * 192 + h * 24;

  const int Hs[NLVL] = {64, 32, 16};
  const int Ws[NLVL] = {192, 96, 48};
  int start = 0;
  float a = 0.f;
#pragma unroll
  for (int l = 0; l < NLVL; ++l) {
    const int Hl = Hs[l], Wl = Ws[l];
#pragma unroll
    for (int p = 0; p < NPT; ++p) {
      const float ox = O[(l * 4 + p) * 2 + 0];
      const float oy = O[(l * 4 + p) * 2 + 1];
      // loc in [0,1]; x = loc*W - 0.5 (align_corners=False, zero padding)
      const float x = (rx + ox / (float)Wl) * (float)Wl - 0.5f;
      const float y = (ry + oy / (float)Hl) * (float)Hl - 0.5f;
      const float x0f = floorf(x), y0f = floorf(y);
      const float wx = x - x0f, wy = y - y0f;
      const int x0 = (int)x0f, y0 = (int)y0f;
      const float w = lg[l * 4 + p] * inv;
      float s = 0.f;
#pragma unroll
      for (int ty = 0; ty < 2; ++ty) {
#pragma unroll
        for (int tx = 0; tx < 2; ++tx) {
          const int xi = x0 + tx, yi = y0 + ty;
          const float tw = (tx ? wx : 1.f - wx) * (ty ? wy : 1.f - wy);
          if (xi >= 0 && xi < Wl && yi >= 0 && yi < Hl)
            s += tw * vproj[(size_t)(start + yi * Wl + xi) * CDIM + h * DH + d];
        }
      }
      a += w * s;
    }
    start += Hl * Wl;
  }
  acc[(size_t)v * CDIM + t] = a;
}

// ---------------- kernel 6: out-proj (WMMA) + identity + LN + mask scatter -
// One wave per 16 voxels: acc fragments loaded ONCE, swept over all 8 column
// tiles of Wout, epilogue (bias + identity) staged in LDS, then layernorm and
// masked channel-major scatter by the same wave.
__global__ __launch_bounds__(32)
void vpl_outproj_ln_scatter(const float* __restrict__ acc,
                            const __bf16* __restrict__ wt_out,
                            const float* __restrict__ bout,
                            const float* __restrict__ se,
                            const float* __restrict__ ln_g,
                            const float* __restrict__ ln_b,
                            const int* __restrict__ mask,
                            const unsigned char* __restrict__ fov,
                            float* __restrict__ out) {
  __shared__ float sm[16][CDIM];
  __shared__ float smean[16], srstd[16];
  const int row0 = blockIdx.x * 16;

  v16bf af[4];
  load_a_frags<false>(af, acc, nullptr, row0);
  {
    const int lane = threadIdx.x & 31;
    const int half = lane >> 4;
    const int n    = lane & 15;
#pragma unroll
    for (int t = 0; t < 8; ++t) {
      const int col0 = t * 16;
      const v8f c = wmma_cols(af, wt_out, col0);
      const float bv = bout[col0 + n];
#pragma unroll
      for (int r = 0; r < 8; ++r) {
        const int m = r + 8 * half;
        sm[m][col0 + n] =
            c[r] + bv + se[(size_t)(row0 + m) * CDIM + col0 + n];
      }
    }
  }
  __syncthreads();

  // per-voxel layernorm stats (population variance, eps = 1e-5)
  if (threadIdx.x < 16) {
    const int m = threadIdx.x;
    float s = 0.f, s2 = 0.f;
    for (int j = 0; j < CDIM; ++j) { const float x = sm[m][j]; s += x; s2 += x * x; }
    const float mean = s * (1.f / CDIM);
    const float var  = s2 * (1.f / CDIM) - mean * mean;
    smean[m] = mean;
    srstd[m] = rsqrtf(var + 1e-5f);
  }
  __syncthreads();

  // masked channel-major scatter: 16 consecutive lanes share a channel ->
  // 64B contiguous stores into out[c*V + v].
  for (int j = 0; j < 64; ++j) {
    const int idx = threadIdx.x + 32 * j;  // 0..2047
    const int cch = idx >> 4;
    const int m   = idx & 15;
    const int v   = row0 + m;
    const float ln =
        (sm[m][cch] - smean[m]) * srstd[m] * ln_g[cch] + ln_b[cch];
    const bool pm = (mask[v] != 0);
    const bool fv = (fov[v] != 0);
    const float o = pm ? (fv ? ln : 0.f) : se[(size_t)v * CDIM + cch];
    out[(size_t)cch * V_TOT + v] = o;
  }
}

// ---------------- launch ----------------------------------------------------
extern "C" void kernel_launch(void* const* d_in, const int* in_sizes, int n_in,
                              void* d_out, int out_size, void* d_ws,
                              size_t ws_size, hipStream_t stream) {
  const float* se    = (const float*)d_in[0];   // scene_embed [V,128]
  const float* sp    = (const float*)d_in[1];   // scene_pos   [V,128]
  const float* ref   = (const float*)d_in[2];   // ref_pix     [V,2]
  const float* f0    = (const float*)d_in[3];   // [128,64,192]
  const float* f1    = (const float*)d_in[4];   // [128,32,96]
  const float* f2    = (const float*)d_in[5];   // [128,16,48]
  const float* Wv    = (const float*)d_in[6];
  const float* bv    = (const float*)d_in[7];
  const float* Woff  = (const float*)d_in[8];
  const float* boff  = (const float*)d_in[9];
  const float* Wattn = (const float*)d_in[10];
  const float* battn = (const float*)d_in[11];
  const float* Wout  = (const float*)d_in[12];
  const float* bout  = (const float*)d_in[13];
  const float* ln_g  = (const float*)d_in[14];
  const float* ln_b  = (const float*)d_in[15];
  const int*   vpts  = (const int*)d_in[16];    // vol_pts [V,3]
  const unsigned char* fov = (const unsigned char*)d_in[17];  // bool [V]

  // workspace partition (~455 MB)
  float*  ws_value = (float*)d_ws;                          // S*128
  float*  ws_vproj = ws_value + (size_t)S_TOT * CDIM;       // S*128
  float*  ws_off   = ws_vproj + (size_t)S_TOT * CDIM;       // V*192
  float*  ws_logit = ws_off   + (size_t)V_TOT * 192;        // V*96
  float*  ws_acc   = ws_logit + (size_t)V_TOT * 96;         // V*128
  int*    ws_mask  = (int*)(ws_acc + (size_t)V_TOT * CDIM); // V ints
  __bf16* wt_v     = (__bf16*)(ws_mask + V_TOT);            // 128*128 bf16
  __bf16* wt_off   = wt_v    + 128 * CDIM;                  // 192*128 bf16
  __bf16* wt_attn  = wt_off  + 192 * CDIM;                  //  96*128 bf16
  __bf16* wt_out   = wt_attn +  96 * CDIM;                  // 128*128 bf16

  // 0) pts_mask
  vpl_mask_init<<<(V_TOT + 255) / 256, 256, 0, stream>>>(ws_mask);
  vpl_mask_scatter<<<(V_TOT + 255) / 256, 256, 0, stream>>>(vpts, ws_mask);

  // 1) value concat + weight transposition/conversion (tiny)
  vpl_build_value<<<(S_TOT * CDIM + 255) / 256, 256, 0, stream>>>(f0, f1, f2,
                                                                  ws_value);
  vpl_weight_bf16<<<(128 * CDIM + 255) / 256, 256, 0, stream>>>(Wv, wt_v, 128);
  vpl_weight_bf16<<<(192 * CDIM + 255) / 256, 256, 0, stream>>>(Woff, wt_off, 192);
  vpl_weight_bf16<<<(96 * CDIM + 255) / 256, 256, 0, stream>>>(Wattn, wt_attn, 96);
  vpl_weight_bf16<<<(128 * CDIM + 255) / 256, 256, 0, stream>>>(Wout, wt_out, 128);

  // 2) value projection: [S,128] @ [128,128] + bv  (vproj laid out [S][head][16])
  {
    dim3 g(S_TOT / 16, CDIM / 16);
    vpl_gemm_vproj<<<g, 32, 0, stream>>>(ws_value, wt_v, bv, ws_vproj);
  }
  // 3) fused offset + attention-logit GEMMs (single pass over q = se+sp)
  vpl_gemm_off_attn<<<V_TOT / 16, 32, 0, stream>>>(se, sp, wt_off, boff,
                                                   wt_attn, battn, ws_off,
                                                   ws_logit);
  // 4) softmax + bilinear sampling
  vpl_sample<<<V_TOT, NHEAD * DH, 0, stream>>>(ref, ws_off, ws_logit, ws_vproj,
                                               ws_acc);
  // 5) out-proj + identity + layernorm + masked channel-major scatter
  vpl_outproj_ln_scatter<<<V_TOT / 16, 32, 0, stream>>>(
      ws_acc, wt_out, bout, se, ln_g, ln_b, ws_mask, fov, (float*)d_out);
}
